// get_model_33311766348247
// MI455X (gfx1250) — matrix operand order repaired and checked by reading, not compiled, not run on hardware
//
#include <hip/hip_runtime.h>

// PointNet++ MSG forward for MI455X (gfx1250, wave32, WMMA).
// GEMM core: v_wmma_f32_16x16x32_f16, BN folded into f16 weights, f32 accum.
// Weight tile staged to LDS via global_load_async_to_lds_b128 (ASYNCcnt).
// A fragments double-buffered so global loads overlap WMMA issue.
// Workspace requirement: ~150 MB.

typedef __attribute__((ext_vector_type(16))) _Float16 v16h;
typedef __attribute__((ext_vector_type(8)))  _Float16 v8h;
typedef __attribute__((ext_vector_type(8)))  float    v8f;

#define EPS_BN 1e-5f

// ---------------- input prep: (B,4,N) -> coords (B,N,3), feats (B,N,1) -----
__global__ void k_prep(const float* __restrict__ xyz, float* __restrict__ coords,
                       float* __restrict__ feats, int B, int N) {
  int gid = blockIdx.x * blockDim.x + threadIdx.x;
  if (gid >= B * N) return;
  int b = gid / N, n = gid - b * N;
  const float* px = xyz + (size_t)b * 4 * N;
  coords[(size_t)gid * 3 + 0] = px[0 * N + n];
  coords[(size_t)gid * 3 + 1] = px[1 * N + n];
  coords[(size_t)gid * 3 + 2] = px[2 * N + n];
  feats[gid] = px[3 * N + n];
}

// ---------------- farthest point sampling: one block per batch -------------
__global__ __launch_bounds__(256) void k_fps(const float* __restrict__ coords,
                                             int N, int npoint, int* __restrict__ fidx) {
  int b = blockIdx.x;
  int t = threadIdx.x;
  __shared__ float sdist[4096];
  __shared__ float rv[256];
  __shared__ int   ri[256];
  __shared__ int   s_cur;
  const float* pc = coords + (size_t)b * N * 3;
  for (int n = t; n < N; n += 256) sdist[n] = 1e10f;
  if (t == 0) s_cur = 0;
  __syncthreads();
  for (int it = 0; it < npoint; ++it) {
    int cur = s_cur;
    if (t == 0) fidx[b * npoint + it] = cur;
    float cx = pc[cur * 3 + 0], cy = pc[cur * 3 + 1], cz = pc[cur * 3 + 2];
    float best = -1.f; int bi = 0;
    for (int n = t; n < N; n += 256) {
      float dx = pc[n * 3 + 0] - cx;
      float dy = pc[n * 3 + 1] - cy;
      float dz = pc[n * 3 + 2] - cz;
      float d = dx * dx + dy * dy + dz * dz;
      float m = fminf(sdist[n], d);
      sdist[n] = m;
      if (m > best) { best = m; bi = n; }
    }
    rv[t] = best; ri[t] = bi;
    __syncthreads();
    for (int off = 128; off > 0; off >>= 1) {
      if (t < off) {
        if (rv[t + off] > rv[t] ||
            (rv[t + off] == rv[t] && ri[t + off] < ri[t])) {
          rv[t] = rv[t + off]; ri[t] = ri[t + off];
        }
      }
      __syncthreads();
    }
    if (t == 0) s_cur = ri[0];
    __syncthreads();
  }
}

// ---------------- gather sampled centroids ---------------------------------
__global__ void k_gather(const float* __restrict__ coords, const int* __restrict__ fidx,
                         float* __restrict__ nx, int Np, int S, int B) {
  int gid = blockIdx.x * blockDim.x + threadIdx.x;
  if (gid >= B * S) return;
  int b = gid / S;
  int id = fidx[gid];
  const float* src = coords + ((size_t)b * Np + id) * 3;
  nx[(size_t)gid * 3 + 0] = src[0];
  nx[(size_t)gid * 3 + 1] = src[1];
  nx[(size_t)gid * 3 + 2] = src[2];
}

// ---------------- ball query: one wave32 per (b,s), ordered ballot append --
__global__ __launch_bounds__(256) void k_ballquery(
    const float* __restrict__ coords, const float* __restrict__ nx,
    int B, int Np, int S, int K, float r2, int* __restrict__ idx) {
  int wid = (blockIdx.x * blockDim.x + threadIdx.x) >> 5;
  int lane = threadIdx.x & 31;
  if (wid >= B * S) return;
  int b = wid / S, s = wid - b * S;
  const float* q = nx + ((size_t)b * S + s) * 3;
  float qx = q[0], qy = q[1], qz = q[2];
  const float* pc = coords + (size_t)b * Np * 3;
  int* out = idx + ((size_t)b * S + s) * K;
  int count = 0;
  int firstn = 0x7fffffff;
  for (int base = 0; base < Np; base += 32) {
    if (count >= K) break;
    int n = base + lane;
    bool in = false;
    if (n < Np) {
      float dx = pc[n * 3 + 0] - qx;
      float dy = pc[n * 3 + 1] - qy;
      float dz = pc[n * 3 + 2] - qz;
      in = (dx * dx + dy * dy + dz * dz) <= r2;
    }
    unsigned mask = (unsigned)__ballot(in ? 1 : 0);
    if (in) {
      int pos = count + __popc(mask & ((1u << lane) - 1u));
      if (pos == 0) firstn = n;
      if (pos < K) out[pos] = n;
    }
    count += __popc(mask);
  }
  if (count > K) count = K;
  for (int off = 16; off > 0; off >>= 1)
    firstn = min(firstn, __shfl_xor(firstn, off));
  for (int p = count + lane; p < K; p += 32) out[p] = firstn;
}

// ---------------- grouping: gather + recenter + zero-pad, fp32 -> f16 ------
__global__ void k_group(const float* __restrict__ coords, const float* __restrict__ points,
                        const float* __restrict__ nx, const int* __restrict__ idx,
                        _Float16* __restrict__ g, int B, int Np, int S, int K,
                        int Cp, int Kp, int s0, int Sc) {
  size_t gid = (size_t)blockIdx.x * blockDim.x + threadIdx.x;
  size_t total = (size_t)B * Sc * K * Kp;
  if (gid >= total) return;
  int c = (int)(gid % Kp);
  size_t row = gid / Kp;
  int k = (int)(row % K);
  size_t r2 = row / K;
  int sl = (int)(r2 % Sc);
  int b = (int)(r2 / Sc);
  int s = s0 + sl;
  int id = idx[((size_t)(b * S + s)) * K + k];
  float v;
  if (c < Cp) {
    v = points[((size_t)b * Np + id) * Cp + c];
  } else if (c < Cp + 3) {
    int d = c - Cp;
    v = coords[((size_t)b * Np + id) * 3 + d] - nx[((size_t)(b * S + s)) * 3 + d];
  } else {
    v = 0.f;
  }
  g[gid] = (_Float16)v;
}

// ---------------- SA3 grouping: [l2_xyz | l2_p] pad to 672 -----------------
__global__ void k_group_sa3(const float* __restrict__ l2xyz, const float* __restrict__ l2p,
                            _Float16* __restrict__ g, int B, int Kp) {
  int gid = blockIdx.x * blockDim.x + threadIdx.x;
  int rows = B * 128;
  if (gid >= rows * Kp) return;
  int c = gid % Kp;
  int row = gid / Kp;
  float v;
  if (c < 3)        v = l2xyz[(size_t)row * 3 + c];
  else if (c < 643) v = l2p[(size_t)row * 640 + (c - 3)];
  else              v = 0.f;
  g[gid] = (_Float16)v;
}

// ---------------- fold BN into f16 transposed weights ----------------------
// Wt[cout][cin_pad] = W[cin][cout] * gamma/sqrt(var+eps); bias folds b,mean,beta.
__global__ void k_fold(const float* __restrict__ W, const float* __restrict__ bvec,
                       const float* __restrict__ gamma, const float* __restrict__ beta,
                       const float* __restrict__ mean, const float* __restrict__ var,
                       int Cin, int Cout, int Kp, int Nt,
                       _Float16* __restrict__ Wt, float* __restrict__ bias) {
  int gid = blockIdx.x * blockDim.x + threadIdx.x;
  if (gid >= Nt * Kp) return;
  int co = gid / Kp, ci = gid - co * Kp;
  if (co >= Cout) {
    Wt[gid] = (_Float16)0.f;
    if (ci == 0) bias[co] = 0.f;
    return;
  }
  float s = 1.f;
  if (gamma) s = gamma[co] * rsqrtf(var[co] + EPS_BN);
  float w = (ci < Cin) ? W[(size_t)ci * Cout + co] : 0.f;
  Wt[gid] = (_Float16)(w * s);
  if (ci == 0) {
    float bb = bvec[co];
    bias[co] = gamma ? ((bb - mean[co]) * s + beta[co]) : bb;
  }
}

// ---------------- WMMA GEMM: Out = act(A @ W + bias) -----------------------
// Block = 8 waves, owns one 16-col N tile and 32 M tiles (4 per wave).
// B tile (16 x Kp) is async-DMA'd to LDS once per block (ASYNCcnt), then each
// wave runs 4 independent accumulator chains sharing one LDS B fragment per
// K-step. A fragments are double-buffered: next K-step's 8 global loads are
// issued before the current 4 WMMAs, so load latency hides under matrix math.
// M indices are clamped (never early-return) so EXEC stays all-ones for WMMA
// and every wave reaches the barrier; stores are guarded.
__global__ __launch_bounds__(256) void k_gemm_wmma(
    const _Float16* __restrict__ A, const _Float16* __restrict__ Wt,
    const float* __restrict__ bias, _Float16* __restrict__ Out,
    int rows, int Kp, int Nt, int relu) {
  extern __shared__ _Float16 lb[];     // 16 x (Kp + 8) halves, padded rows
  const int KpL = Kp + 8;
  int tiles_n = Nt >> 4, tiles_m = rows >> 4;
  int tn = blockIdx.x % tiles_n;
  int mb = blockIdx.x / tiles_n;
  int tid = threadIdx.x;
  int w = tid >> 5, lane = tid & 31;
  int hl = lane & 15;   // M row in tile (A) / N col in tile (B)
  int sel = lane >> 4;  // K-half selector

  // --- stage B tile (16 cols x Kp halves) into LDS via async DMA ---
  {
    int chunks = Kp >> 3;  // 16-byte units per column
    const _Float16* gbase = Wt + (size_t)tn * 16 * Kp;
    for (int u = tid; u < 16 * chunks; u += 256) {
      int col = u / chunks, ch = u - col * chunks;
      unsigned long long src =
          (unsigned long long)(size_t)(gbase + (size_t)col * Kp + ch * 8);
      unsigned dst = (unsigned)(size_t)(lb + col * KpL + ch * 8);
      asm volatile("global_load_async_to_lds_b128 %0, %1, off"
                   :: "v"(dst), "v"(src) : "memory");
    }
    asm volatile("s_wait_asynccnt 0" ::: "memory");
  }
  __syncthreads();

  // --- 4 M-subtiles per wave, clamped in-range ---
  int mt0 = (mb * 8 + w) * 4;
  const _Float16* ar[4];
#pragma unroll
  for (int i = 0; i < 4; ++i) {
    int mt = mt0 + i;
    int mtc = mt < tiles_m ? mt : tiles_m - 1;
    ar[i] = A + (size_t)(mtc * 16 + hl) * Kp + sel * 8;
  }
  const _Float16* bl = lb + hl * KpL + sel * 8;

  v8f zero = {0.f, 0.f, 0.f, 0.f, 0.f, 0.f, 0.f, 0.f};
  v8f acc[4];
#pragma unroll
  for (int i = 0; i < 4; ++i) acc[i] = zero;

  // prologue: load K-step 0 fragments
  v16h acur[4];
#pragma unroll
  for (int i = 0; i < 4; ++i) {
    v8h alo = *(const v8h*)(ar[i]);
    v8h ahi = *(const v8h*)(ar[i] + 16);
#pragma unroll
    for (int j = 0; j < 8; ++j) { acur[i][j] = alo[j]; acur[i][j + 8] = ahi[j]; }
  }

  for (int k0 = 0; k0 < Kp; k0 += 32) {
    // issue next K-step's A loads first (clamped tail keeps addresses valid)
    int kn = (k0 + 32 < Kp) ? (k0 + 32) : k0;
    v16h anext[4];
#pragma unroll
    for (int i = 0; i < 4; ++i) {
      v8h alo = *(const v8h*)(ar[i] + kn);
      v8h ahi = *(const v8h*)(ar[i] + kn + 16);
#pragma unroll
      for (int j = 0; j < 8; ++j) { anext[i][j] = alo[j]; anext[i][j + 8] = ahi[j]; }
    }
    // B fragment from LDS (shared by all 4 chains)
    v8h blo = *(const v8h*)(bl + k0);
    v8h bhi = *(const v8h*)(bl + k0 + 16);
    v16h bm;
#pragma unroll
    for (int j = 0; j < 8; ++j) { bm[j] = blo[j]; bm[j + 8] = bhi[j]; }
    // 4 independent WMMA chains on current fragments
#pragma unroll
    for (int i = 0; i < 4; ++i)
      acc[i] = __builtin_amdgcn_wmma_f32_16x16x32_f16(false, acur[i], false, bm,
                                                      (short)0, acc[i], false, false);
#pragma unroll
    for (int i = 0; i < 4; ++i) acur[i] = anext[i];
  }

  // C/D layout: VGPR r -> M = r + sel*8, N = hl
  int ncol = tn * 16 + hl;
  float bb = bias[ncol];
#pragma unroll
  for (int i = 0; i < 4; ++i) {
    int mt = mt0 + i;
    if (mt < tiles_m) {
      int mbase = mt * 16 + sel * 8;
#pragma unroll
      for (int r = 0; r < 8; ++r) {
        float v = acc[i][r] + bb;
        if (relu) v = fmaxf(v, 0.f);
        Out[(size_t)(mbase + r) * Nt + ncol] = (_Float16)v;
      }
    }
  }
}

// ---------------- max-pool over K neighbors, write fp32 into concat --------
__global__ void k_maxpool(const _Float16* __restrict__ in, float* __restrict__ out,
                          int B, int Sc, int K, int C, int Nt, int S, int s0,
                          int Ctot, int coff) {
  int gid = blockIdx.x * blockDim.x + threadIdx.x;
  if (gid >= B * Sc * C) return;
  int c = gid % C;
  int r = gid / C;
  int sl = r % Sc, b = r / Sc;
  const _Float16* p = in + ((size_t)(b * Sc + sl) * K) * Nt + c;
  float m = -3.4e38f;
  for (int k = 0; k < K; ++k) {
    float v = (float)p[(size_t)k * Nt];
    m = fmaxf(m, v);
  }
  out[((size_t)b * S + s0 + sl) * Ctot + coff + c] = m;
}

// ---------------- SA3 max-pool: l3 fp32 to d_out and f16 for FC ------------
__global__ void k_maxpool_sa3(const _Float16* __restrict__ in, float* __restrict__ l3out,
                              _Float16* __restrict__ l3h, int B, int J, int C) {
  int gid = blockIdx.x * blockDim.x + threadIdx.x;
  if (gid >= B * C) return;
  int b = gid / C, c = gid - b * C;
  const _Float16* p = in + ((size_t)b * J) * C + c;
  float m = -3.4e38f;
  for (int j = 0; j < J; ++j) {
    float v = (float)p[(size_t)j * C];
    m = fmaxf(m, v);
  }
  l3out[gid] = m;
  l3h[gid] = (_Float16)m;
}

// ---------------- log-softmax: one wave per row ----------------------------
__global__ void k_logsoftmax(const _Float16* __restrict__ logits, float* __restrict__ out,
                             int C, int stride) {
  int b = blockIdx.x;
  int lane = threadIdx.x;
  float mx = -3.4e38f;
  for (int c = lane; c < C; c += 32) mx = fmaxf(mx, (float)logits[b * stride + c]);
  for (int off = 16; off > 0; off >>= 1) mx = fmaxf(mx, __shfl_xor(mx, off));
  float se = 0.f;
  for (int c = lane; c < C; c += 32) se += expf((float)logits[b * stride + c] - mx);
  for (int off = 16; off > 0; off >>= 1) se += __shfl_xor(se, off);
  float ls = logf(se);
  for (int c = lane; c < C; c += 32)
    out[b * C + c] = (float)logits[b * stride + c] - mx - ls;
}

// ===========================================================================
extern "C" void kernel_launch(void* const* d_in, const int* in_sizes, int n_in,
                              void* d_out, int out_size, void* d_ws, size_t ws_size,
                              hipStream_t stream) {
  (void)in_sizes; (void)n_in; (void)out_size; (void)ws_size;
  const int B = 16, N = 4096;
  auto F = [&](int i) -> const float* { return (const float*)d_in[i]; };

  // --- workspace allocator (256B aligned) ---
  char* wsb = (char*)d_ws;
  size_t off = 0;
  auto alloc = [&](size_t bytes) -> void* {
    void* p = wsb + off;
    off = (off + bytes + 255) & ~(size_t)255;
    return p;
  };

  float*    coords = (float*)alloc((size_t)B * N * 3 * 4);
  float*    feats  = (float*)alloc((size_t)B * N * 4);
  int*      fidx1  = (int*)alloc((size_t)B * 512 * 4);
  float*    l1xyz  = (float*)alloc((size_t)B * 512 * 3 * 4);
  float*    l1p    = (float*)alloc((size_t)B * 512 * 320 * 4);
  int*      fidx2  = (int*)alloc((size_t)B * 128 * 4);
  float*    l2xyz  = (float*)alloc((size_t)B * 128 * 3 * 4);
  float*    l2p    = (float*)alloc((size_t)B * 128 * 640 * 4);
  int*      idxbuf = (int*)alloc((size_t)B * 512 * 128 * 4);
  _Float16* bufA   = (_Float16*)alloc((size_t)25165824 * 2);  // ping (~48 MB)
  _Float16* bufB   = (_Float16*)alloc((size_t)33554432 * 2);  // pong (~64 MB)
  _Float16* l3h    = (_Float16*)alloc((size_t)B * 1024 * 2);
  _Float16* fco1   = (_Float16*)alloc((size_t)16 * 512 * 2);
  _Float16* fco2   = (_Float16*)alloc((size_t)16 * 256 * 2);
  _Float16* fco3   = (_Float16*)alloc((size_t)16 * 48 * 2);

  struct LW { _Float16* Wt; float* bias; int Kp, Nt, Cout; };
  auto pad16 = [](int x) { return (x + 15) & ~15; };
  auto pad32 = [](int x) { return (x + 31) & ~31; };

  auto fold = [&](int iW, int ib, int igamma, int ibeta, int imean, int ivar,
                  int Cin, int Cout) -> LW {
    LW L;
    L.Kp = pad32(Cin); L.Nt = pad16(Cout); L.Cout = Cout;
    L.Wt = (_Float16*)alloc((size_t)L.Nt * L.Kp * 2);
    L.bias = (float*)alloc((size_t)L.Nt * 4);
    int tot = L.Nt * L.Kp;
    k_fold<<<(tot + 255) / 256, 256, 0, stream>>>(
        F(iW), F(ib),
        igamma >= 0 ? F(igamma) : nullptr, ibeta >= 0 ? F(ibeta) : nullptr,
        imean >= 0 ? F(imean) : nullptr, ivar >= 0 ? F(ivar) : nullptr,
        Cin, Cout, L.Kp, L.Nt, L.Wt, L.bias);
    return L;
  };

  // Param flat order (jax pytree, sorted dict keys):
  // 0:xyz, 1-4:bn1(beta,gamma,mean,var), 5-8:bn2, 9-10:fc1(W,b), 11-12:fc2,
  // 13-14:fc3, then sa1/sa2/sa3 layers each as (W,b,beta,gamma,mean,var).
  const int SA1 = 15, SA2 = 69, SA3 = 123;
  auto mk = [&](int basep, int scale, int layer, int Cin, int Cout) -> LW {
    int p = basep + scale * 18 + layer * 6;
    return fold(p + 0, p + 1, p + 3, p + 2, p + 4, p + 5, Cin, Cout);
  };

  int sa1_spec[3][4] = {{4, 32, 32, 64}, {4, 64, 64, 128}, {4, 64, 96, 128}};
  int sa2_spec[3][4] = {{323, 64, 64, 128}, {323, 128, 128, 256}, {323, 128, 128, 256}};
  int sa3_spec[4] = {643, 256, 512, 1024};
  LW Lsa1[3][3], Lsa2[3][3], Lsa3[3];
  for (int s = 0; s < 3; s++)
    for (int l = 0; l < 3; l++)
      Lsa1[s][l] = mk(SA1, s, l, sa1_spec[s][l], sa1_spec[s][l + 1]);
  for (int s = 0; s < 3; s++)
    for (int l = 0; l < 3; l++)
      Lsa2[s][l] = mk(SA2, s, l, sa2_spec[s][l], sa2_spec[s][l + 1]);
  for (int l = 0; l < 3; l++)
    Lsa3[l] = mk(SA3, 0, l, sa3_spec[l], sa3_spec[l + 1]);
  LW Lfc1 = fold(9, 10, 2, 1, 3, 4, 1024, 512);
  LW Lfc2 = fold(11, 12, 6, 5, 7, 8, 512, 256);
  LW Lfc3 = fold(13, 14, -1, -1, -1, -1, 256, 40);

  auto gemm = [&](const _Float16* A, const LW& L, _Float16* Out, int rows, int relu) {
    int tiles_m = rows / 16, tiles_n = L.Nt / 16;
    int mbs = (tiles_m + 31) / 32;           // 32 M-tiles per block
    size_t shmem = (size_t)16 * (L.Kp + 8) * 2;
    k_gemm_wmma<<<tiles_n * mbs, 256, shmem, stream>>>(A, L.Wt, L.bias, Out,
                                                       rows, L.Kp, L.Nt, relu);
  };

  // --- prep ---
  k_prep<<<(B * N + 255) / 256, 256, 0, stream>>>(F(0), coords, feats, B, N);

  // --- generic set-abstraction MSG stage, chunked over S for L2 residency ---
  auto sa_stage = [&](const float* cc, int Np, const float* pts, int Cp,
                      int S, int Sc, int* fidx, float* lxyz, float* lp, int Ctot,
                      const float* radii, const int* Ks, LW (*Ls)[3]) {
    k_fps<<<B, 256, 0, stream>>>(cc, Np, S, fidx);
    k_gather<<<(B * S + 255) / 256, 256, 0, stream>>>(cc, fidx, lxyz, Np, S, B);
    int coff = 0;
    for (int sc = 0; sc < 3; sc++) {
      int K = Ks[sc];
      float r = radii[sc];
      int waves = B * S;
      k_ballquery<<<(waves * 32 + 255) / 256, 256, 0, stream>>>(
          cc, lxyz, B, Np, S, K, r * r, idxbuf);
      int C = Ls[sc][2].Cout;
      for (int s0 = 0; s0 < S; s0 += Sc) {
        int rows = B * Sc * K;
        int Kp = Ls[sc][0].Kp;
        size_t tot = (size_t)rows * Kp;
        k_group<<<(unsigned)((tot + 255) / 256), 256, 0, stream>>>(
            cc, pts, lxyz, idxbuf, bufA, B, Np, S, K, Cp, Kp, s0, Sc);
        gemm(bufA, Ls[sc][0], bufB, rows, 1);
        gemm(bufB, Ls[sc][1], bufA, rows, 1);
        gemm(bufA, Ls[sc][2], bufB, rows, 1);
        k_maxpool<<<(B * Sc * C + 255) / 256, 256, 0, stream>>>(
            bufB, lp, B, Sc, K, C, Ls[sc][2].Nt, S, s0, Ctot, coff);
      }
      coff += C;
    }
  };

  float radii1[3] = {0.1f, 0.2f, 0.4f}; int K1[3] = {16, 32, 128};
  sa_stage(coords, N, feats, 1, 512, 128, fidx1, l1xyz, l1p, 320, radii1, K1, Lsa1);

  float radii2[3] = {0.2f, 0.4f, 0.8f}; int K2[3] = {32, 64, 128};
  sa_stage(l1xyz, 512, l1p, 320, 128, 32, fidx2, l2xyz, l2p, 640, radii2, K2, Lsa2);

  // --- SA3 (group-all) ---
  {
    int rows = B * 128;
    int Kp = Lsa3[0].Kp;  // 672
    k_group_sa3<<<(rows * Kp + 255) / 256, 256, 0, stream>>>(l2xyz, l2p, bufA, B, Kp);
    gemm(bufA, Lsa3[0], bufB, rows, 1);   // 2048 x 256
    gemm(bufB, Lsa3[1], bufA, rows, 1);   // 2048 x 512
    gemm(bufA, Lsa3[2], bufB, rows, 1);   // 2048 x 1024
    float* out = (float*)d_out;
    k_maxpool_sa3<<<(B * 1024 + 255) / 256, 256, 0, stream>>>(
        bufB, out + 640, l3h, B, 128, 1024);
  }

  // --- FC head ---
  gemm(l3h, Lfc1, fco1, 16, 1);
  gemm(fco1, Lfc2, fco2, 16, 1);
  gemm(fco2, Lfc3, fco3, 16, 0);
  k_logsoftmax<<<16, 32, 0, stream>>>(fco3, (float*)d_out, 40, 48);
}
